// PointGCN_28475633173131
// MI455X (gfx1250) — compile-verified, matrix operand order checked by reading, not run on hardware
//
#include <hip/hip_runtime.h>
#include <hip/hip_bf16.h>

// ---------------- problem constants ----------------
constexpr int NB = 4;            // batch
constexpr int NP = 4096;         // points
constexpr int KS = 16;           // knn k
constexpr int M1 = NB * NP;      // 16384 point rows
constexpr int ME = M1 * KS;      // 262144 edge rows
constexpr int KPAD_MAX = 1152;   // largest padded K (W11)
constexpr float BN_EPS = 1e-5f;

// ---------------- WMMA / TDM types ----------------
typedef __attribute__((ext_vector_type(16))) __bf16 v16bf;
typedef __attribute__((ext_vector_type(8)))  float  v8f;
typedef unsigned int u32x4 __attribute__((ext_vector_type(4)));
typedef int          i32x8 __attribute__((ext_vector_type(8)));
typedef int          i32x4 __attribute__((ext_vector_type(4)));

union FragBF { unsigned int u[8]; v16bf v; };

__device__ inline unsigned short f2bf(float f) {
    union { float f; unsigned int u; } x; x.f = f;
    unsigned int u = x.u;
    unsigned int r = (u + 0x7FFFu + ((u >> 16) & 1u)) >> 16;   // RNE
    return (unsigned short)r;
}

// ======================================================================
// KNN: brute-force top-16 smallest d2 = |p|^2 + |q|^2 - 2 p.q (stable ties)
// ======================================================================
__global__ void __launch_bounds__(256)
k_knn(const float* __restrict__ data, int* __restrict__ idx)
{
    const int b = blockIdx.y;
    const int n = blockIdx.x * 256 + threadIdx.x;
    const float* base = data + (size_t)b * NP * 9;

    const float px = base[n * 9 + 0];
    const float py = base[n * 9 + 1];
    const float pz = base[n * 9 + 2];
    const float sp = px * px + py * py + pz * pz;

    __shared__ float sx[256], sy[256], sz[256], sq[256];

    float bd[KS]; int bi[KS];
#pragma unroll
    for (int k = 0; k < KS; ++k) { bd[k] = 3.4e38f; bi[k] = 0; }

    for (int t = 0; t < NP; t += 256) {
        const int m = t + threadIdx.x;
        const float qx = base[m * 9 + 0];
        const float qy = base[m * 9 + 1];
        const float qz = base[m * 9 + 2];
        sx[threadIdx.x] = qx; sy[threadIdx.x] = qy; sz[threadIdx.x] = qz;
        sq[threadIdx.x] = qx * qx + qy * qy + qz * qz;
        __syncthreads();
        for (int j = 0; j < 256; ++j) {
            const float d = sp + sq[j] - 2.0f * (px * sx[j] + py * sy[j] + pz * sz[j]);
            if (d < bd[KS - 1]) {
                int p = KS - 1;
                while (p > 0 && bd[p - 1] > d) { bd[p] = bd[p - 1]; bi[p] = bi[p - 1]; --p; }
                bd[p] = d; bi[p] = t + j;
            }
        }
        __syncthreads();
    }
    int* o = idx + ((size_t)(b * NP + n)) * KS;
#pragma unroll
    for (int k = 0; k < KS; ++k) o[k] = bi[k];
}

// ======================================================================
// f32 [M, C] (row stride ldSrc) -> bf16 [M, Kpad] zero padded
// ======================================================================
__global__ void k_f32_to_bf16pad(const float* __restrict__ src, unsigned short* __restrict__ dst,
                                 int M, int C, int ldSrc, int Kpad)
{
    size_t gid = (size_t)blockIdx.x * 256 + threadIdx.x;
    size_t total = (size_t)M * Kpad;
    if (gid >= total) return;
    int r = (int)(gid / Kpad);
    int c = (int)(gid % Kpad);
    float v = (c < C) ? src[(size_t)r * ldSrc + c] : 0.0f;
    dst[gid] = f2bf(v);
}

// W [K, Nc] row-major f32 -> Wt bf16 [NcPad, Kpad] zero padded (transposed)
__global__ void k_wt(const float* __restrict__ W, unsigned short* __restrict__ Wt,
                     int Kact, int Nc, int Kpad, int NcPad)
{
    size_t gid = (size_t)blockIdx.x * 256 + threadIdx.x;
    size_t total = (size_t)NcPad * Kpad;
    if (gid >= total) return;
    int nc = (int)(gid / Kpad);
    int k  = (int)(gid % Kpad);
    float v = (nc < Nc && k < Kact) ? W[(size_t)k * Nc + nc] : 0.0f;
    Wt[gid] = f2bf(v);
}

// ======================================================================
// WMMA GEMM: C[M,NcPad] = A[M,Kpad](bf16) x Bt[NcPad,Kpad](bf16, N-major)
//
// Block = up to 4 waves. The block's 16-row A panel (16 x Kpad bf16) is
// DMA'd to LDS once by the Tensor Data Mover (TENSOR_LOAD_TO_LDS,
// TENSORcnt), then each wave computes NT 16x16 output tiles
// (nTile = blockIdx.y*4*NT + j*4 + wave), reusing one A fragment (read
// from LDS) across NT back-to-back WMMAs per K-step.
// NT is a template parameter so the inner loop is branch-free and EXEC is
// statically all-ones at every WMMA.
// ======================================================================
template <int NT>
__global__ void __launch_bounds__(128)
k_gemm_bf16(const unsigned short* __restrict__ A,
            const unsigned short* __restrict__ Bt,
            float* __restrict__ C,
            int Kpad, int NcPad)
{
    __shared__ unsigned short shA[16 * KPAD_MAX];

    const int wave = threadIdx.x >> 5;
    const int lane = threadIdx.x & 31;
    const int mTile = blockIdx.x;

    // ---- TDM: load A panel [16, Kpad] (contiguous) into LDS, wave 0 only ----
    if (threadIdx.x < 32) {
        const unsigned long long ga =
            (unsigned long long)(uintptr_t)(A + (size_t)mTile * 16 * Kpad);
        const unsigned int lds = (unsigned int)(uintptr_t)&shA[0]; // low 32 bits = LDS offset

        // D# group 0: count=1, lds_addr, global_addr, type=2
        u32x4 g0;
        g0[0] = 1u;                                   // count=1, is_restore=0, gather=0
        g0[1] = lds;                                  // lds_addr
        g0[2] = (unsigned int)(ga & 0xFFFFFFFFull);   // global_addr[31:0]
        g0[3] = (unsigned int)((ga >> 32) & 0x1FFFFFFull) | (2u << 30); // ga[56:32], type=2

        // D# group 1: data_size=2B, 2-D tile [Kpad x 16], stride = Kpad
        const unsigned int td0 = (unsigned int)Kpad;  // tensor_dim0 (elements)
        const unsigned int td1 = 0x7FFFFFFFu;         // tensor_dim1: huge (no OOB clip)
        const unsigned int tile0 = (unsigned int)Kpad & 0xFFFFu;
        const unsigned int tile1 = 16u;
        const unsigned long long s0 = (unsigned long long)Kpad; // tensor_dim0_stride
        i32x8 g1;
        g1[0] = (int)(1u << 16);                                   // data_size=1 (2 bytes)
        g1[1] = (int)((td0 & 0xFFFFu) << 16);                      // tensor_dim0[15:0]
        g1[2] = (int)((td0 >> 16) | ((td1 & 0xFFFFu) << 16));      // td0 hi | td1 lo
        g1[3] = (int)((td1 >> 16) | (tile0 << 16));                // td1 hi | tile_dim0
        g1[4] = (int)tile1;                                        // tile_dim1 (tile_dim2=0)
        g1[5] = (int)(unsigned int)(s0 & 0xFFFFFFFFull);           // stride0[31:0]
        g1[6] = (int)(unsigned int)((s0 >> 32) & 0xFFFFull);       // stride0[47:32]
        g1[7] = 0;                                                 // stride1 hi
        i32x4 g2 = {0, 0, 0, 0};
        i32x4 g3 = {0, 0, 0, 0};
        i32x8 g4 = {0, 0, 0, 0, 0, 0, 0, 0};                       // 6-arg toolchain extra group

        __builtin_amdgcn_tensor_load_to_lds(g0, g1, g2, g3, g4, 0);
        __builtin_amdgcn_s_wait_tensorcnt(0);
    }
    __syncthreads();

    const int half16 = lane >> 4;                    // 0/1
    const int sub    = lane & 15;

    // NT tiles per wave, all statically valid
    int nTile[NT];
    const unsigned short* bRow[NT];
#pragma unroll
    for (int j = 0; j < NT; ++j) {
        nTile[j] = blockIdx.y * (4 * NT) + j * 4 + wave;
        bRow[j]  = Bt + (size_t)(nTile[j] * 16 + sub) * Kpad;
    }

    const unsigned short* aRow = &shA[0] + (size_t)sub * Kpad;

    v8f acc[NT] = {};
    for (int k0 = 0; k0 < Kpad; k0 += 32) {
        FragBF a;
#pragma unroll
        for (int i = 0; i < 8; ++i) {
            const int k = k0 + ((i >> 2) * 16) + (half16 * 8) + ((i & 3) * 2);
            a.u[i] = *(const unsigned int*)(aRow + k);        // LDS read
        }
#pragma unroll
        for (int j = 0; j < NT; ++j) {
            FragBF b;
#pragma unroll
            for (int i = 0; i < 8; ++i) {
                const int k = k0 + ((i >> 2) * 16) + (half16 * 8) + ((i & 3) * 2);
                b.u[i] = *(const unsigned int*)(bRow[j] + k);
            }
            acc[j] = __builtin_amdgcn_wmma_f32_16x16x32_bf16(
                         false, a.v, false, b.v, (short)0, acc[j], false, false);
        }
    }

    const int rowBase = mTile * 16 + half16 * 8;
#pragma unroll
    for (int j = 0; j < NT; ++j) {
        const int col = nTile[j] * 16 + sub;
#pragma unroll
        for (int v = 0; v < 8; ++v)
            C[(size_t)(rowBase + v) * NcPad + col] = acc[j][v];
    }
}

// ======================================================================
// BatchNorm (global over leading axes): reduce -> stats -> apply(+relu)
// ======================================================================
__global__ void k_zero2(float* a, float* b, int n)
{
    int i = blockIdx.x * 256 + threadIdx.x;
    if (i < n) { a[i] = 0.0f; b[i] = 0.0f; }
}

__global__ void __launch_bounds__(256)
k_bn_reduce(const float* __restrict__ X, float* __restrict__ sums, float* __restrict__ sq,
            size_t total, int C)
{
    __shared__ float ssum[1024];
    __shared__ float ssq[1024];
    for (int c = threadIdx.x; c < C; c += 256) { ssum[c] = 0.0f; ssq[c] = 0.0f; }
    __syncthreads();

    const size_t stride = (size_t)gridDim.x * 256;
    for (size_t e = (size_t)blockIdx.x * 256 + threadIdx.x; e < total; e += stride) {
        const float v = X[e];
        const int c = (int)(e & (size_t)(C - 1));   // C is a power of two
        atomicAdd(&ssum[c], v);
        atomicAdd(&ssq[c], v * v);
    }
    __syncthreads();
    for (int c = threadIdx.x; c < C; c += 256) {
        atomicAdd(&sums[c], ssum[c]);
        atomicAdd(&sq[c],   ssq[c]);
    }
}

__global__ void k_bn_stats(const float* __restrict__ sums, const float* __restrict__ sq,
                           float* __restrict__ mean, float* __restrict__ inv, int C, float invM)
{
    int c = blockIdx.x * 256 + threadIdx.x;
    if (c >= C) return;
    float m = sums[c] * invM;
    float v = sq[c] * invM - m * m;
    mean[c] = m;
    inv[c]  = rsqrtf(v + BN_EPS);
}

__global__ void k_bn_apply(const float* __restrict__ X, float* __restrict__ Y,
                           const float* __restrict__ mean, const float* __restrict__ inv,
                           size_t total, int Cmask)
{
    size_t gid = (size_t)blockIdx.x * 256 + threadIdx.x;
    if (gid >= total) return;
    int c = (int)(gid & (size_t)Cmask);
    float v = (X[gid] - mean[c]) * inv[c];
    Y[gid] = v > 0.0f ? v : 0.0f;
}

// bn+relu then max over the K (=16) neighbor axis: H[(row*16+k), C] -> Y[row, C]
__global__ void k_bn_max_apply(const float* __restrict__ H, float* __restrict__ Y,
                               const float* __restrict__ mean, const float* __restrict__ inv,
                               int Mrows, int C)
{
    size_t gid = (size_t)blockIdx.x * 256 + threadIdx.x;
    size_t total = (size_t)Mrows * C;
    if (gid >= total) return;
    int row = (int)(gid / C);
    int c   = (int)(gid % C);
    const float m = mean[c], s = inv[c];
    float best = -3.4e38f;
    const float* h = H + (size_t)row * KS * C + c;
#pragma unroll
    for (int k = 0; k < KS; ++k) {
        float v = (h[(size_t)k * C] - m) * s;
        v = v > 0.0f ? v : 0.0f;
        best = v > best ? v : best;
    }
    Y[gid] = best;
}

// ======================================================================
// graph ops
// ======================================================================
__global__ void k_gather_max(const float* __restrict__ F, const int* __restrict__ idx,
                             float* __restrict__ Y, int C)
{
    size_t gid = (size_t)blockIdx.x * 256 + threadIdx.x;
    size_t total = (size_t)M1 * C;
    if (gid >= total) return;
    int row = (int)(gid / C);
    int c   = (int)(gid % C);
    int b   = row / NP;
    const int* ii = idx + (size_t)row * KS;
    float best = -3.4e38f;
#pragma unroll
    for (int k = 0; k < KS; ++k) {
        float v = F[((size_t)(b * NP + ii[k])) * C + c];
        best = v > best ? v : best;
    }
    Y[gid] = best;
}

// edge features 1: [x0(64) | x0[nb]-x0(64) | pts[nb]-pts(3) | pad..160] -> bf16
__global__ void k_build_edge1(const float* __restrict__ x0, const float* __restrict__ data,
                              const int* __restrict__ idx, unsigned short* __restrict__ E)
{
    size_t gid = (size_t)blockIdx.x * 256 + threadIdx.x;
    size_t total = (size_t)ME * 160;
    if (gid >= total) return;
    int row = (int)(gid / 160);          // (b*NP+n)*KS + k
    int c   = (int)(gid % 160);
    int bn  = row / KS;
    int b   = bn / NP;
    int nb  = idx[row];
    float v;
    if (c < 64)        v = x0[(size_t)bn * 64 + c];
    else if (c < 128)  v = x0[((size_t)(b * NP + nb)) * 64 + (c - 64)] - x0[(size_t)bn * 64 + (c - 64)];
    else if (c < 131)  { int j = c - 128;
                         v = data[((size_t)(b * NP + nb)) * 9 + j] - data[(size_t)bn * 9 + j]; }
    else               v = 0.0f;
    E[gid] = f2bf(v);
}

// edge features 2: [x(64) | x[nb]-x(64)] -> bf16
__global__ void k_build_edge2(const float* __restrict__ x, const int* __restrict__ idx,
                              unsigned short* __restrict__ E)
{
    size_t gid = (size_t)blockIdx.x * 256 + threadIdx.x;
    size_t total = (size_t)ME * 128;
    if (gid >= total) return;
    int row = (int)(gid / 128);
    int c   = (int)(gid % 128);
    int bn  = row / KS;
    int b   = bn / NP;
    int nb  = idx[row];
    float v;
    if (c < 64) v = x[(size_t)bn * 64 + c];
    else        v = x[((size_t)(b * NP + nb)) * 64 + (c - 64)] - x[(size_t)bn * 64 + (c - 64)];
    E[gid] = f2bf(v);
}

// x12 = [x1 | x2] -> bf16 [M1, 128]
__global__ void k_concat2(const float* __restrict__ x1, const float* __restrict__ x2,
                          unsigned short* __restrict__ E)
{
    size_t gid = (size_t)blockIdx.x * 256 + threadIdx.x;
    size_t total = (size_t)M1 * 128;
    if (gid >= total) return;
    int r = (int)(gid / 128);
    int c = (int)(gid % 128);
    float v = (c < 64) ? x1[(size_t)r * 64 + c] : x2[(size_t)r * 64 + (c - 64)];
    E[gid] = f2bf(v);
}

// g[b,c] = max_n H[b,n,c]  (C = 1024)
__global__ void __launch_bounds__(256)
k_max_n(const float* __restrict__ H, float* __restrict__ g, int C)
{
    int c = blockIdx.x * 256 + threadIdx.x;
    int b = blockIdx.y;
    if (c >= C) return;
    float best = -3.4e38f;
    for (int n = 0; n < NP; ++n) {
        float v = H[((size_t)(b * NP + n)) * C + c];
        best = v > best ? v : best;
    }
    g[(size_t)b * C + c] = best;
}

// cat = [g(1024) | x1(64) | x2(64)] -> bf16 [M1, 1152]
__global__ void k_build_cat(const float* __restrict__ g, const float* __restrict__ x1,
                            const float* __restrict__ x2, unsigned short* __restrict__ E)
{
    size_t gid = (size_t)blockIdx.x * 256 + threadIdx.x;
    size_t total = (size_t)M1 * 1152;
    if (gid >= total) return;
    int r = (int)(gid / 1152);
    int c = (int)(gid % 1152);
    int b = r / NP;
    float v;
    if (c < 1024)      v = g[(size_t)b * 1024 + c];
    else if (c < 1088) v = x1[(size_t)r * 64 + (c - 1024)];
    else               v = x2[(size_t)r * 64 + (c - 1088)];
    E[gid] = f2bf(v);
}

// H[(b*NP+n)*16 + c] (c<13) -> out[b,13,NP]
__global__ void k_transpose_out(const float* __restrict__ H, float* __restrict__ out)
{
    size_t gid = (size_t)blockIdx.x * 256 + threadIdx.x;
    size_t total = (size_t)NB * 13 * NP;
    if (gid >= total) return;
    int n = (int)(gid % NP);
    int c = (int)((gid / NP) % 13);
    int b = (int)(gid / ((size_t)13 * NP));
    out[gid] = H[((size_t)(b * NP + n)) * 16 + c];
}

// ======================================================================
// host orchestration
// ======================================================================
extern "C" void kernel_launch(void* const* d_in, const int* in_sizes, int n_in,
                              void* d_out, int out_size, void* d_ws, size_t ws_size,
                              hipStream_t stream)
{
    (void)in_sizes; (void)n_in; (void)out_size; (void)ws_size;

    const float* data = (const float*)d_in[0];
    const float* W_fc = (const float*)d_in[1];
    // d_in[2] = b_fc: constant per-channel bias cancels exactly under BN mean-subtraction.
    const float* W1  = (const float*)d_in[3];
    const float* W2  = (const float*)d_in[4];
    const float* W3  = (const float*)d_in[5];
    const float* W4  = (const float*)d_in[6];
    const float* W5  = (const float*)d_in[7];
    const float* W6  = (const float*)d_in[8];
    const float* W10 = (const float*)d_in[9];
    const float* W11 = (const float*)d_in[10];
    const float* W12 = (const float*)d_in[11];
    const float* W13 = (const float*)d_in[12];

    char* ws = (char*)d_ws;
    size_t off = 0;
    auto alloc = [&](size_t bytes) -> void* {
        void* p = ws + off;
        off += (bytes + 255) & ~(size_t)255;
        return p;
    };

    int*   IDX  = (int*)  alloc((size_t)M1 * KS * 4);
    float* X0   = (float*)alloc((size_t)M1 * 64  * 4);
    float* XC   = (float*)alloc((size_t)M1 * 256 * 4);
    float* XA   = (float*)alloc((size_t)M1 * 256 * 4);
    float* X1   = (float*)alloc((size_t)M1 * 64  * 4);
    float* X2   = (float*)alloc((size_t)M1 * 64  * 4);
    float* G    = (float*)alloc((size_t)NB * 1024 * 4);
    float* SUM  = (float*)alloc(1024 * 4);
    float* SQ   = (float*)alloc(1024 * 4);
    float* MEAN = (float*)alloc(1024 * 4);
    float* INV  = (float*)alloc(1024 * 4);
    unsigned short* ABUF = (unsigned short*)alloc((size_t)ME * 160 * 2);
    float*          HBUF = (float*)         alloc((size_t)ME * 256 * 4);

    auto wt = [&](int ncp, int kp) { return (unsigned short*)alloc((size_t)ncp * kp * 2); };
    unsigned short* WT0  = wt(64, 32);
    unsigned short* WT1  = wt(256, 160);
    unsigned short* WT2  = wt(64, 256);
    unsigned short* WT3  = wt(64, 64);
    unsigned short* WT4  = wt(256, 128);
    unsigned short* WT5  = wt(64, 256);
    unsigned short* WT6  = wt(64, 64);
    unsigned short* WT10 = wt(1024, 128);
    unsigned short* WT11 = wt(512, 1152);
    unsigned short* WT12 = wt(256, 512);
    unsigned short* WT13 = wt(16, 256);

    auto blocks = [](size_t total) { return dim3((unsigned)((total + 255) / 256)); };

    auto wtrans = [&](const float* W, unsigned short* Wt, int Kact, int Nc, int Kpad, int NcPad) {
        k_wt<<<blocks((size_t)NcPad * Kpad), 256, 0, stream>>>(W, Wt, Kact, Nc, Kpad, NcPad);
    };
    auto gemm = [&](const unsigned short* A, const unsigned short* Bt, float* C,
                    int M, int Kpad, int NcPad) {
        if (NcPad % 256 == 0) {
            // 4 waves x 4 tiles: block covers 16 N-tiles (256 cols)
            dim3 grid(M / 16, (unsigned)(NcPad / 256));
            k_gemm_bf16<4><<<grid, 128, 0, stream>>>(A, Bt, C, Kpad, NcPad);
        } else if (NcPad == 64) {
            // 4 waves x 1 tile
            dim3 grid(M / 16, 1);
            k_gemm_bf16<1><<<grid, 128, 0, stream>>>(A, Bt, C, Kpad, NcPad);
        } else { // NcPad == 16: single wave, single tile
            dim3 grid(M / 16, 1);
            k_gemm_bf16<1><<<grid, 32, 0, stream>>>(A, Bt, C, Kpad, NcPad);
        }
    };
    auto bn = [&](float* X, size_t Mrows, int C) {
        k_zero2<<<blocks(1024), 256, 0, stream>>>(SUM, SQ, 1024);
        k_bn_reduce<<<dim3(2048), 256, 0, stream>>>(X, SUM, SQ, Mrows * (size_t)C, C);
        k_bn_stats<<<blocks(1024), 256, 0, stream>>>(SUM, SQ, MEAN, INV, C, 1.0f / (float)Mrows);
    };

    // 1) knn indices from xyz
    k_knn<<<dim3(NP / 256, NB), 256, 0, stream>>>(data, IDX);

    // 2) x0 = bn_relu(data @ W_fc)    (bias cancels under BN)
    k_f32_to_bf16pad<<<blocks((size_t)M1 * 32), 256, 0, stream>>>(data, ABUF, M1, 9, 9, 32);
    wtrans(W_fc, WT0, 9, 64, 32, 64);
    gemm(ABUF, WT0, HBUF, M1, 32, 64);
    bn(HBUF, M1, 64);
    k_bn_apply<<<blocks((size_t)M1 * 64), 256, 0, stream>>>(HBUF, X0, MEAN, INV, (size_t)M1 * 64, 63);

    // 3) edge_conv1 (with pts diff), W1: 131->256
    k_build_edge1<<<blocks((size_t)ME * 160), 256, 0, stream>>>(X0, data, IDX, ABUF);
    wtrans(W1, WT1, 131, 256, 160, 256);
    gemm(ABUF, WT1, HBUF, ME, 160, 256);
    bn(HBUF, (size_t)ME, 256);
    k_bn_max_apply<<<blocks((size_t)M1 * 256), 256, 0, stream>>>(HBUF, XC, MEAN, INV, M1, 256);

    // 4) graph_conv W2: 256->64
    k_gather_max<<<blocks((size_t)M1 * 256), 256, 0, stream>>>(XC, IDX, XA, 256);
    k_f32_to_bf16pad<<<blocks((size_t)M1 * 256), 256, 0, stream>>>(XA, ABUF, M1, 256, 256, 256);
    wtrans(W2, WT2, 256, 64, 256, 64);
    gemm(ABUF, WT2, HBUF, M1, 256, 64);
    bn(HBUF, M1, 64);
    k_bn_apply<<<blocks((size_t)M1 * 64), 256, 0, stream>>>(HBUF, XC, MEAN, INV, (size_t)M1 * 64, 63);

    // 5) graph_conv W3: 64->64  -> x1
    k_gather_max<<<blocks((size_t)M1 * 64), 256, 0, stream>>>(XC, IDX, XA, 64);
    k_f32_to_bf16pad<<<blocks((size_t)M1 * 64), 256, 0, stream>>>(XA, ABUF, M1, 64, 64, 64);
    wtrans(W3, WT3, 64, 64, 64, 64);
    gemm(ABUF, WT3, HBUF, M1, 64, 64);
    bn(HBUF, M1, 64);
    k_bn_apply<<<blocks((size_t)M1 * 64), 256, 0, stream>>>(HBUF, X1, MEAN, INV, (size_t)M1 * 64, 63);

    // 6) edge_conv2 (no pts), W4: 128->256
    k_build_edge2<<<blocks((size_t)ME * 128), 256, 0, stream>>>(X1, IDX, ABUF);
    wtrans(W4, WT4, 128, 256, 128, 256);
    gemm(ABUF, WT4, HBUF, ME, 128, 256);
    bn(HBUF, (size_t)ME, 256);
    k_bn_max_apply<<<blocks((size_t)M1 * 256), 256, 0, stream>>>(HBUF, XC, MEAN, INV, M1, 256);

    // 7) graph_conv W5: 256->64
    k_gather_max<<<blocks((size_t)M1 * 256), 256, 0, stream>>>(XC, IDX, XA, 256);
    k_f32_to_bf16pad<<<blocks((size_t)M1 * 256), 256, 0, stream>>>(XA, ABUF, M1, 256, 256, 256);
    wtrans(W5, WT5, 256, 64, 256, 64);
    gemm(ABUF, WT5, HBUF, M1, 256, 64);
    bn(HBUF, M1, 64);
    k_bn_apply<<<blocks((size_t)M1 * 64), 256, 0, stream>>>(HBUF, XC, MEAN, INV, (size_t)M1 * 64, 63);

    // 8) graph_conv W6: 64->64 -> x2
    k_gather_max<<<blocks((size_t)M1 * 64), 256, 0, stream>>>(XC, IDX, XA, 64);
    k_f32_to_bf16pad<<<blocks((size_t)M1 * 64), 256, 0, stream>>>(XA, ABUF, M1, 64, 64, 64);
    wtrans(W6, WT6, 64, 64, 64, 64);
    gemm(ABUF, WT6, HBUF, M1, 64, 64);
    bn(HBUF, M1, 64);
    k_bn_apply<<<blocks((size_t)M1 * 64), 256, 0, stream>>>(HBUF, X2, MEAN, INV, (size_t)M1 * 64, 63);

    // 9) h = bn_relu([x1|x2] @ W10); g = max_n h
    k_concat2<<<blocks((size_t)M1 * 128), 256, 0, stream>>>(X1, X2, ABUF);
    wtrans(W10, WT10, 128, 1024, 128, 1024);
    gemm(ABUF, WT10, HBUF, M1, 128, 1024);
    bn(HBUF, M1, 1024);
    k_bn_apply<<<blocks((size_t)M1 * 1024), 256, 0, stream>>>(HBUF, HBUF, MEAN, INV, (size_t)M1 * 1024, 1023);
    k_max_n<<<dim3(1024 / 256, NB), 256, 0, stream>>>(HBUF, G, 1024);

    // 10) head: cat -> W11 -> W12 -> W13 -> transpose
    k_build_cat<<<blocks((size_t)M1 * 1152), 256, 0, stream>>>(G, X1, X2, ABUF);
    wtrans(W11, WT11, 1152, 512, 1152, 512);
    gemm(ABUF, WT11, HBUF, M1, 1152, 512);
    bn(HBUF, M1, 512);
    k_bn_apply<<<blocks((size_t)M1 * 512), 256, 0, stream>>>(HBUF, HBUF, MEAN, INV, (size_t)M1 * 512, 511);

    k_f32_to_bf16pad<<<blocks((size_t)M1 * 512), 256, 0, stream>>>(HBUF, ABUF, M1, 512, 512, 512);
    wtrans(W12, WT12, 512, 256, 512, 256);
    gemm(ABUF, WT12, HBUF, M1, 512, 256);
    bn(HBUF, M1, 256);
    k_bn_apply<<<blocks((size_t)M1 * 256), 256, 0, stream>>>(HBUF, HBUF, MEAN, INV, (size_t)M1 * 256, 255);

    k_f32_to_bf16pad<<<blocks((size_t)M1 * 256), 256, 0, stream>>>(HBUF, ABUF, M1, 256, 256, 256);
    wtrans(W13, WT13, 256, 13, 256, 16);
    gemm(ABUF, WT13, HBUF, M1, 256, 16);

    k_transpose_out<<<blocks((size_t)NB * 13 * NP), 256, 0, stream>>>(HBUF, (float*)d_out);
}
